// GatedLinearAttention_32332513804900
// MI455X (gfx1250) — matrix operand order repaired and checked by reading, not compiled, not run on hardware
//
#include <hip/hip_runtime.h>
#include <hip/hip_bf16.h>
#include <math.h>

typedef __attribute__((ext_vector_type(16))) _Float16 v16h;
typedef __attribute__((ext_vector_type(8)))  _Float16 v8h;
typedef __attribute__((ext_vector_type(4)))  _Float16 h4;
typedef __attribute__((ext_vector_type(8)))  float    v8f;
typedef __attribute__((ext_vector_type(4)))  float    f4;

#define WMMA_F16(a,b,c) __builtin_amdgcn_wmma_f32_16x16x32_f16(false,(a),false,(b),(short)0,(c),false,false)

// Problem constants (B=2, L=2048)
#define ROWS   4096
#define DMODEL 1024
#define NHEAD  4
#define DK     256
#define DV     512
#define DVS    128       // Dv slice per workgroup in the scan
#define CHUNK  32
#define LSEQ   2048

// padded LDS strides (floats) for the scan: keep 16B alignment + spread banks
#define LQ  260          // qs/ks row stride [t][dk]
#define LS  260          // S row stride [n][dk]
#define LT  36           // ksT/vsh/Amat row stride (k dim = 32)

// load 8 contiguous floats, convert to halfs into d[o..o+7]
__device__ __forceinline__ void cvt8(v16h& d, int o, const float* p) {
    f4 a = *(const f4*)p;
    f4 b = *(const f4*)(p + 4);
    #pragma unroll
    for (int j = 0; j < 4; ++j) { d[o + j] = (_Float16)a[j]; d[o + 4 + j] = (_Float16)b[j]; }
}
// copy 8 contiguous halfs into d[o..o+7]
__device__ __forceinline__ void cp8(v16h& d, int o, const _Float16* p) {
    v8h t = *(const v8h*)p;
    #pragma unroll
    for (int j = 0; j < 8; ++j) d[o + j] = t[j];
}

// ---------------------------------------------------------------------------
// WMMA GEMM: C = alpha * A(MxK,f32) @ B(KxN,f32). Tile 128x128x32, 8 waves,
// wave tile 32x64 (2x4 of 16x16). A staged [m][k], B staged transposed [n][k]
// so every fragment load is contiguous ds_load_b128s. M%128==0, K%32==0.
// FULLN=true instantiation is fully branch-free (all big calls); the N=16
// low-rank call uses FULLN=false.
// ---------------------------------------------------------------------------
#define LDA 40   // halfs per Ash row
#define LDB 40   // halfs per Bsht row

template <bool FULLN>
__global__ __launch_bounds__(256) void gemm_f16(const float* __restrict__ A,
                                                const float* __restrict__ B,
                                                float* __restrict__ C,
                                                int M, int N, int K, float alpha)
{
    __shared__ _Float16 Ash[128 * LDA];    // [m][k]
    __shared__ _Float16 Bsht[128 * LDB];   // [n][k] (transposed)

    const int tid  = threadIdx.x;
    const int lane = tid & 31;
    const int w    = tid >> 5;
    const int wm   = w & 3;                // 32-row slice
    const int wn   = w >> 2;               // 64-col slice
    const int m0   = blockIdx.y * 128;
    const int n0   = blockIdx.x * 128;

    v8f acc[2][4] = {};

    for (int k0 = 0; k0 < K; k0 += 32) {
        // prefetch next K-tile while this one is staged/consumed
        if (k0 + 32 < K) {
            __builtin_prefetch(A + (size_t)(m0 + (tid >> 3)) * K + (k0 + 32) + (tid & 7) * 4, 0, 3);
            __builtin_prefetch(B + (size_t)(k0 + 32 + (tid >> 3)) * N + n0 + (tid & 7) * 16, 0, 3);
        }
        // stage A: 128x32, 4 rows/thread, float4 coalesced loads, 8B half stores
        {
            const int cg = (tid & 7) * 4;
            #pragma unroll
            for (int i = 0; i < 4; ++i) {
                const int row = (tid >> 3) + 32 * i;
                f4 v = *(const f4*)(A + (size_t)(m0 + row) * K + k0 + cg);
                h4 hv;
                #pragma unroll
                for (int j = 0; j < 4; ++j) hv[j] = (_Float16)v[j];
                *(h4*)&Ash[row * LDA + cg] = hv;
            }
        }
        // stage B: 32x128 -> transposed [n][k]
        {
            const int row = tid >> 3;            // k row 0..31
            const int cgb = (tid & 7) * 16;      // n group
            if constexpr (FULLN) {
                const float* src = B + (size_t)(k0 + row) * N + n0 + cgb;
                #pragma unroll
                for (int q = 0; q < 4; ++q) {
                    f4 v = *(const f4*)(src + 4 * q);
                    #pragma unroll
                    for (int j = 0; j < 4; ++j)
                        Bsht[(cgb + 4 * q + j) * LDB + row] = (_Float16)v[j];
                }
            } else {
                #pragma unroll
                for (int j = 0; j < 16; ++j) {
                    const int n = n0 + cgb + j;
                    const float v = (n < N) ? B[(size_t)(k0 + row) * N + n] : 0.f;
                    Bsht[(cgb + j) * LDB + row] = (_Float16)v;
                }
            }
        }
        __syncthreads();

        // fragments: all contiguous b128 LDS loads
        v16h af[2], bf[4];
        {
            const int r    = lane & 15;
            const int kb8  = (lane >> 4) * 8;
            const int kb16 = (lane >> 4) * 16;
            #pragma unroll
            for (int mt = 0; mt < 2; ++mt) {
                const _Float16* base = &Ash[(wm * 32 + mt * 16 + r) * LDA];
                cp8(af[mt], 0, base + kb16);
                cp8(af[mt], 8, base + kb16 + 8);
            }
            #pragma unroll
            for (int nt = 0; nt < 4; ++nt) {
                const _Float16* base = &Bsht[(wn * 64 + nt * 16 + r) * LDB + kb16];
                cp8(bf[nt], 0, base);
                cp8(bf[nt], 8, base + 8);
            }
        }
        #pragma unroll
        for (int mt = 0; mt < 2; ++mt)
            #pragma unroll
            for (int nt = 0; nt < 4; ++nt)
                acc[mt][nt] = WMMA_F16(af[mt], bf[nt], acc[mt][nt]);
        __syncthreads();
    }

    const int nco = n0 + wn * 64 + (lane & 15);
    const int mro = m0 + wm * 32 + ((lane >> 4) << 3);
    #pragma unroll
    for (int mt = 0; mt < 2; ++mt)
        #pragma unroll
        for (int nt = 0; nt < 4; ++nt)
            #pragma unroll
            for (int r = 0; r < 8; ++r) {
                const int m = mro + mt * 16 + r;
                const int n = nco + nt * 16;
                if (FULLN || n < N) C[(size_t)m * N + n] = alpha * acc[mt][nt][r];
            }
}

// ---------------------------------------------------------------------------
// gk = log_sigmoid(A1 @ gk_w2 + b2) / 16 ;  A1: [ROWS,16], gk_w2: [16,1024]
// ---------------------------------------------------------------------------
__global__ __launch_bounds__(256) void gk_lowrank(const float* __restrict__ A1,
                                                  const float* __restrict__ w2,
                                                  const float* __restrict__ b2,
                                                  float* __restrict__ GK)
{
    const int idx = blockIdx.x * 256 + threadIdx.x;
    const int row = idx >> 10;
    const int col = idx & 1023;
    float s = b2[col];
    #pragma unroll
    for (int j = 0; j < 16; ++j) s += A1[row * 16 + j] * w2[j * 1024 + col];
    const float ls = fminf(s, 0.f) - log1pf(__expf(-fabsf(s)));
    GK[idx] = ls * (1.f / 16.f);
}

// ---------------------------------------------------------------------------
// Chunked GLA scan. 32 blocks (b,h,DvSlice), 256 threads = 8 waves; each wave
// owns 16 Dv columns of the state (wave-private). All WMMA operands are laid
// out in LDS so fragment loads are contiguous vector loads.
//   S   [n][dk]  (LS stride)   state, f32
//   qs  [t][dk]  (LQ)          q * exp(b)
//   ks  [t][dk]  (LQ)          k * exp(-b)      (B operand of scores)
//   ksT [dk][t]  (LT)          k * exp(bC - b)  (A operand of state update)
//   vsh [n][t]   (LT)          v chunk slice    (B operand, transposed)
//   Amat[t][s]   (LT)          masked scores
// ---------------------------------------------------------------------------
__global__ __launch_bounds__(256) void gla_scan(const float* __restrict__ Q,
                                                const float* __restrict__ Kp,
                                                const float* __restrict__ V,
                                                const float* __restrict__ GK,
                                                float* __restrict__ O)
{
    extern __shared__ float lds[];
    float* S     = lds;                    // [DVS][LS]
    float* qs    = S    + DVS * LS;        // [CHUNK][LQ]
    float* ks    = qs   + CHUNK * LQ;      // [CHUNK][LQ]
    float* ksT   = ks   + CHUNK * LQ;      // [DK][LT]
    float* Amat  = ksT  + DK * LT;         // [CHUNK][LT]
    float* vsh   = Amat + CHUNK * LT;      // [DVS][LT]
    float* dlast = vsh  + DVS * LT;        // [DK]

    const int tid  = threadIdx.x;
    const int lane = tid & 31;
    const int w    = tid >> 5;
    const int bid  = blockIdx.x;
    const int bb   = bid >> 4;
    const int h    = (bid >> 2) & 3;
    const int dv0  = (bid & 3) * DVS;

    const size_t row0 = (size_t)bb * LSEQ;
    const int qcol = h * DK;
    const int vcol = h * DV + dv0;
    const int nloc = w * 16;

    for (int i = tid; i < DVS * LS; i += 256) S[i] = 0.f;
    __syncthreads();

    for (int c = 0; c < LSEQ / CHUNK; ++c) {
        const int l0 = c * CHUNK;

        // -- step 1: per-channel cumulative decay (channel dk = tid)
        {
            const int dk = tid;
            float kloc[CHUNK];
            float bsum = 0.f;
            #pragma unroll
            for (int t = 0; t < CHUNK; ++t) {
                const size_t off = (row0 + l0 + t) * (size_t)DMODEL + qcol + dk;
                bsum += GK[off];
                qs[t * LQ + dk] = Q[off] * __expf(bsum);
                const float kv = Kp[off] * __expf(-bsum);
                ks[t * LQ + dk] = kv;
                kloc[t] = kv;
            }
            const float dl = __expf(bsum);
            dlast[dk] = dl;
            #pragma unroll
            for (int t = 0; t < CHUNK; ++t) ksT[dk * LT + t] = kloc[t] * dl;
        }
        __syncthreads();

        // stage v chunk slice transposed: vsh[n][t]  (coalesced global reads)
        {
            const int t  = tid >> 3;
            const int cg = (tid & 7) * 16;
            const float* src = V + (row0 + l0 + t) * (size_t)(NHEAD * DV) + vcol + cg;
            #pragma unroll
            for (int q = 0; q < 4; ++q) {
                f4 v = *(const f4*)(src + 4 * q);
                #pragma unroll
                for (int j = 0; j < 4; ++j) vsh[(cg + 4 * q + j) * LT + t] = v[j];
            }
        }

        // -- step 2: intra scores A = (q~ @ k~^T) with causal mask (waves 0-3)
        if (w < 4) {
            const int mt = w >> 1, nt = w & 1;
            const int r    = lane & 15;
            const int kb8  = (lane >> 4) * 8;
            const int kb16 = (lane >> 4) * 16;
            v8f acc = {};
            #pragma unroll
            for (int kt = 0; kt < 8; ++kt) {
                v16h afr, bfr;
                const float* ab = &qs[(mt * 16 + r) * LQ + kt * 32];
                cvt8(afr, 0, ab + kb16);
                cvt8(afr, 8, ab + kb16 + 8);
                const float* bbp = &ks[(nt * 16 + r) * LQ + kt * 32 + kb16];
                cvt8(bfr, 0, bbp);
                cvt8(bfr, 8, bbp + 8);
                acc = WMMA_F16(afr, bfr, acc);
            }
            const int s = nt * 16 + (lane & 15);
            #pragma unroll
            for (int rr = 0; rr < 8; ++rr) {
                const int t = mt * 16 + rr + ((lane >> 4) << 3);
                Amat[t * LT + s] = (s <= t) ? acc[rr] : 0.f;
            }
        }
        __syncthreads();

        // v fragment (B operand, K = t): contiguous from vsh[n][t]
        v16h vfr;
        {
            const float* vb = &vsh[(nloc + (lane & 15)) * LT + (lane >> 4) * 8];
            cvt8(vfr, 0, vb);
            cvt8(vfr, 8, vb + 16);
        }

        // -- step 3: o = A@v + q~@S_prev   (wave-private n columns)
        {
            const int r    = lane & 15;
            const int kb8  = (lane >> 4) * 8;
            const int kb16 = (lane >> 4) * 16;
            const int n    = lane & 15;
            #pragma unroll
            for (int mt = 0; mt < 2; ++mt) {
                v8f acc = {};
                v16h afr;
                const float* ab = &Amat[(mt * 16 + r) * LT];
                cvt8(afr, 0, ab + kb8);
                cvt8(afr, 8, ab + 16 + kb8);
                acc = WMMA_F16(afr, vfr, acc);                 // intra
                #pragma unroll
                for (int kt = 0; kt < 8; ++kt) {               // inter: q~ @ S
                    v16h qfr, sfr;
                    const float* qb = &qs[(mt * 16 + r) * LQ + kt * 32];
                    cvt8(qfr, 0, qb + kb16);
                    cvt8(qfr, 8, qb + kb16 + 8);
                    const float* sb = &S[(nloc + n) * LS + kt * 32 + kb16];
                    cvt8(sfr, 0, sb);
                    cvt8(sfr, 8, sb + 8);
                    acc = WMMA_F16(qfr, sfr, acc);
                }
                #pragma unroll
                for (int rr = 0; rr < 8; ++rr) {
                    const int t = mt * 16 + rr + ((lane >> 4) << 3);
                    O[(row0 + l0 + t) * (size_t)(NHEAD * DV) + vcol + nloc + n] = acc[rr];
                }
            }
        }

        // -- step 4: S = diag(exp(bC)) S + k̂^T @ v   (contiguous [n][dk] rows)
        {
            const int r   = lane & 15;
            const int kb8 = (lane >> 4) * 8;
            const int n   = lane & 15;
            const int ro8 = (lane >> 4) << 3;
            #pragma unroll
            for (int mt = 0; mt < 16; ++mt) {
                const int dkb = mt * 16 + ro8;
                float* srow = &S[(nloc + n) * LS + dkb];
                const float* dlp = &dlast[dkb];
                v8f cc;
                #pragma unroll
                for (int rr = 0; rr < 8; ++rr) cc[rr] = srow[rr] * dlp[rr];
                v16h afr;                                       // k̂^T, pre-scaled
                const float* ab = &ksT[(mt * 16 + r) * LT];
                cvt8(afr, 0, ab + kb8);
                cvt8(afr, 8, ab + 16 + kb8);
                const v8f dd = WMMA_F16(afr, vfr, cc);
                #pragma unroll
                for (int rr = 0; rr < 8; ++rr) srow[rr] = dd[rr];
            }
        }
        __syncthreads();
    }
}

// ---------------------------------------------------------------------------
// Fused RMSNorm + SwishGate: one block per (row, head), 256 thr, Dv=512.
// ---------------------------------------------------------------------------
__global__ __launch_bounds__(256) void normgate(const float* __restrict__ O,
                                                const float* __restrict__ G,
                                                const float* __restrict__ gw,
                                                float* __restrict__ Y)
{
    __shared__ float red[256];
    const int tid = threadIdx.x;
    const int row = blockIdx.x >> 2;
    const int h   = blockIdx.x & 3;
    const size_t base = (size_t)row * (NHEAD * DV) + h * DV;

    const float o0 = O[base + tid], o1 = O[base + tid + 256];
    red[tid] = o0 * o0 + o1 * o1;
    __syncthreads();
    for (int s = 128; s > 0; s >>= 1) {
        if (tid < s) red[tid] += red[tid + s];
        __syncthreads();
    }
    const float inv = rsqrtf(red[0] * (1.f / 512.f) + 1e-5f);
    const float g0 = G[base + tid], g1 = G[base + tid + 256];
    Y[base + tid]       = o0 * inv * gw[tid]       * (g0 / (1.f + __expf(-g0)));
    Y[base + tid + 256] = o1 * inv * gw[tid + 256] * (g1 / (1.f + __expf(-g1)));
}

// ---------------------------------------------------------------------------
extern "C" void kernel_launch(void* const* d_in, const int* in_sizes, int n_in,
                              void* d_out, int out_size, void* d_ws, size_t ws_size,
                              hipStream_t stream)
{
    const float* x    = (const float*)d_in[0];
    const float* Wq   = (const float*)d_in[1];
    const float* Wk   = (const float*)d_in[2];
    const float* Wv   = (const float*)d_in[3];
    const float* Wg   = (const float*)d_in[4];
    const float* gkw1 = (const float*)d_in[5];
    const float* gkw2 = (const float*)d_in[6];
    const float* gkb2 = (const float*)d_in[7];
    const float* gnw  = (const float*)d_in[8];
    const float* Wo   = (const float*)d_in[9];
    float*       out  = (float*)d_out;

    float* ws  = (float*)d_ws;
    float* Qb  = ws;
    float* Kb  = Qb  + (size_t)ROWS * 1024;
    float* GKb = Kb  + (size_t)ROWS * 1024;
    float* Vb  = GKb + (size_t)ROWS * 1024;
    float* Gb  = Vb  + (size_t)ROWS * 2048;
    float* A1  = Gb  + (size_t)ROWS * 2048;
    float* Ob  = A1  + (size_t)ROWS * 16;
    float* Yb  = Ob  + (size_t)ROWS * 2048;

    const dim3 blk(256);

    gemm_f16<true><<<dim3(1024 / 128, ROWS / 128), blk, 0, stream>>>(x, Wq, Qb, ROWS, 1024, 1024, 0.0625f);
    gemm_f16<true><<<dim3(1024 / 128, ROWS / 128), blk, 0, stream>>>(x, Wk, Kb, ROWS, 1024, 1024, 1.f);
    gemm_f16<true><<<dim3(2048 / 128, ROWS / 128), blk, 0, stream>>>(x, Wv, Vb, ROWS, 2048, 1024, 1.f);
    gemm_f16<true><<<dim3(2048 / 128, ROWS / 128), blk, 0, stream>>>(x, Wg, Gb, ROWS, 2048, 1024, 1.f);
    gemm_f16<false><<<dim3(1,          ROWS / 128), blk, 0, stream>>>(x, gkw1, A1, ROWS, 16, 1024, 1.f);

    gk_lowrank<<<(ROWS * 1024) / 256, blk, 0, stream>>>(A1, gkw2, gkb2, GKb);

    const size_t scan_lds =
        (size_t)(DVS * LS + 2 * CHUNK * LQ + DK * LT + CHUNK * LT + DVS * LT + DK) * sizeof(float);
    gla_scan<<<32, blk, scan_lds, stream>>>(Qb, Kb, Vb, GKb, Ob);

    normgate<<<ROWS * NHEAD, blk, 0, stream>>>(Ob, Gb, gnw, Yb);

    gemm_f16<true><<<dim3(1024 / 128, ROWS / 128), blk, 0, stream>>>(Yb, Wo, out, ROWS, 1024, 2048, 1.f);
}
